// GCN_6502580486349
// MI455X (gfx1250) — compile-verified, matrix operand order checked by reading, not run on hardware
//
#include <hip/hip_runtime.h>

#define NN 100000
#define NE 1600000
#define D  128
#define NG 512
#define NC 4

typedef __bf16 bf16;
typedef __attribute__((ext_vector_type(8)))  __bf16 v8bf;
typedef __attribute__((ext_vector_type(16))) __bf16 v16bf;
typedef __attribute__((ext_vector_type(8)))  float  v8f;

// ---------------------------------------------------------------- utilities

__global__ void zero_f32(float* __restrict__ p, int n) {
  int i = blockIdx.x * blockDim.x + threadIdx.x;
  if (i < n) p[i] = 0.0f;
}

__global__ void f2bf_kernel(const float* __restrict__ in, bf16* __restrict__ out, int n) {
  int i = blockIdx.x * blockDim.x + threadIdx.x;
  if (i < n) out[i] = (bf16)in[i];
}

// ---------------------------------------------------------------- degree

__global__ void deg_count(const int* __restrict__ dst, float* __restrict__ deg) {
  int e = blockIdx.x * blockDim.x + threadIdx.x;
  if (e < NE) atomicAdd(&deg[dst[e]], 1.0f);
}

__global__ void deg_invert(float* __restrict__ deg) {
  int i = blockIdx.x * blockDim.x + threadIdx.x;
  if (i < NN) deg[i] = 1.0f / fmaxf(deg[i], 1.0f);
}

// ------------------------------------------------- edge scatter (L2-resident)
// 32 threads per edge; each gathers float4 of the src row and scatter-adds
// into the dst row with global_atomic_add_f32 (resolves in L2: 51MB << 192MB).

__global__ void scatter_add(const int* __restrict__ src, const int* __restrict__ dst,
                            const float* __restrict__ X, float* __restrict__ agg) {
  long long tid = (long long)blockIdx.x * blockDim.x + threadIdx.x;
  int e = (int)(tid >> 5);
  if (e >= NE) return;
  int c = ((int)tid & 31) * 4;
  int s = src[e], d = dst[e];
  float4 v = *(const float4*)(X + (size_t)s * D + c);
  float* p = agg + (size_t)d * D + c;
  atomicAdd(p + 0, v.x);
  atomicAdd(p + 1, v.y);
  atomicAdd(p + 2, v.z);
  atomicAdd(p + 3, v.w);
}

// scale aggregated sums by 1/deg and convert to bf16 for the WMMA GEMM
__global__ void scale_to_bf16(const float* __restrict__ agg, const float* __restrict__ deg_inv,
                              bf16* __restrict__ out) {
  int i = blockIdx.x * blockDim.x + threadIdx.x;
  if (i < NN * D) out[i] = (bf16)(agg[i] * deg_inv[i >> 7]);
}

// ---------------------------------------------------------------- WMMA GEMM
// Fragment loader. `base` is row-major [*, 128] bf16. 16x32 bf16 A-operand
// layout (ISA 7.12.2): lane L holds row M = L&15; half = L>>4 selects K
// sub-blocks {half*8..half*8+7} and {16+half*8..16+half*8+7}. B uses the
// symmetric layout with the weight row n playing the role of M (B[k][n] =
// W[n][k]), so the same contiguous-row load serves both operands.
__device__ __forceinline__ v16bf load_frag(const bf16* __restrict__ base,
                                           int row0, int k0, int lane) {
  int r = lane & 15, half = lane >> 4;
  const bf16* p = base + (size_t)(row0 + r) * D + k0 + half * 8;
  v8bf lo = *(const v8bf*)p;
  v8bf hi = *(const v8bf*)(p + 16);
  return __builtin_shufflevector(lo, hi, 0, 1, 2, 3, 4, 5, 6, 7,
                                         8, 9, 10, 11, 12, 13, 14, 15);
}

// out[m][n] = relu( Ab@Wl^T + Rb@Wr^T + bias ), M=n_rows, N=K=128.
// One wave per 16-row strip. A-fragments (4 K-chunks x 2 matrices) are hoisted
// into VGPRs; inner loop is 8 N-tiles x (4 K-chunks x 2 WMMA) = 64
// v_wmma_f32_16x16x32_bf16 per wave, weights stream from L1/L2.
__global__ __launch_bounds__(256) void sage_gemm(
    const bf16* __restrict__ Ab, const bf16* __restrict__ Rb,
    const bf16* __restrict__ Wl, const bf16* __restrict__ Wr,
    const float* __restrict__ bias, float* __restrict__ out, int n_rows) {
  int lane = threadIdx.x & 31;
  int wave = blockIdx.x * (blockDim.x >> 5) + (threadIdx.x >> 5);
  int m0 = wave * 16;
  if (m0 >= n_rows) return;  // wave-uniform: EXEC stays all-ones for WMMA

  v16bf a[4], r[4];
#pragma unroll
  for (int k = 0; k < 4; ++k) {
    a[k] = load_frag(Ab, m0, k * 32, lane);
    r[k] = load_frag(Rb, m0, k * 32, lane);
  }

  int cn = lane & 15, ch = lane >> 4;  // C/D: acc[i] = D[ch*8+i][cn]
#pragma unroll
  for (int nt = 0; nt < 8; ++nt) {
    int n0 = nt * 16;
    v8f acc = {};
#pragma unroll
    for (int k = 0; k < 4; ++k) {
      v16bf bl = load_frag(Wl, n0, k * 32, lane);
      acc = __builtin_amdgcn_wmma_f32_16x16x32_bf16(false, a[k], false, bl,
                                                    (short)0, acc, false, false);
      v16bf br = load_frag(Wr, n0, k * 32, lane);
      acc = __builtin_amdgcn_wmma_f32_16x16x32_bf16(false, r[k], false, br,
                                                    (short)0, acc, false, false);
    }
    float b = bias[n0 + cn];
#pragma unroll
    for (int i = 0; i < 8; ++i) {
      float v = acc[i] + b;
      out[(size_t)(m0 + ch * 8 + i) * D + n0 + cn] = fmaxf(v, 0.0f);
    }
  }
}

// ---------------------------------------------------------------- pooling

__global__ void pool_sum(const float* __restrict__ h, const int* __restrict__ batch,
                         float* __restrict__ gsum) {
  long long tid = (long long)blockIdx.x * blockDim.x + threadIdx.x;
  int node = (int)(tid >> 5);
  if (node >= NN) return;
  int c = ((int)tid & 31) * 4;
  int g = batch[node];
  float4 v = *(const float4*)(h + (size_t)node * D + c);
  float* p = gsum + (size_t)g * D + c;
  atomicAdd(p + 0, v.x);
  atomicAdd(p + 1, v.y);
  atomicAdd(p + 2, v.z);
  atomicAdd(p + 3, v.w);
}

__global__ void pool_cnt(const int* __restrict__ batch, float* __restrict__ gcnt) {
  int i = blockIdx.x * blockDim.x + threadIdx.x;
  if (i < NN) atomicAdd(&gcnt[batch[i]], 1.0f);
}

// final head: out[g][c] = (gsum[g]/max(cnt,1)) . Wlin[c] + blin[c]  (512x4)
__global__ void final_lin(const float* __restrict__ gsum, const float* __restrict__ gcnt,
                          const float* __restrict__ Wlin, const float* __restrict__ blin,
                          float* __restrict__ out) {
  int tid = blockIdx.x * blockDim.x + threadIdx.x;
  if (tid >= NG * NC) return;
  int g = tid >> 2, c = tid & 3;
  float inv = 1.0f / fmaxf(gcnt[g], 1.0f);
  float s = 0.0f;
#pragma unroll 4
  for (int k = 0; k < D; ++k) s += gsum[(size_t)g * D + k] * Wlin[(size_t)c * D + k];
  out[(size_t)g * NC + c] = s * inv + blin[c];
}

// ---------------------------------------------------------------- driver

static inline size_t align_up(size_t v) { return (v + 255) & ~(size_t)255; }

extern "C" void kernel_launch(void* const* d_in, const int* in_sizes, int n_in,
                              void* d_out, int out_size, void* d_ws, size_t ws_size,
                              hipStream_t stream) {
  const float* x    = (const float*)d_in[0];
  const int*   edge = (const int*)d_in[1];
  const int*   batch= (const int*)d_in[2];
  const float* W1l  = (const float*)d_in[3];
  const float* b1   = (const float*)d_in[4];
  const float* W1r  = (const float*)d_in[5];
  const float* W2l  = (const float*)d_in[6];
  const float* b2   = (const float*)d_in[7];
  const float* W2r  = (const float*)d_in[8];
  const float* Wlin = (const float*)d_in[9];
  const float* blin = (const float*)d_in[10];
  const int* src = edge;           // edge_index[0]
  const int* dst = edge + NE;      // edge_index[1]
  float* out = (float*)d_out;

  // workspace carve-up
  char* ws = (char*)d_ws;
  size_t off = 0;
  float* deg  = (float*)(ws + off); off = align_up(off + (size_t)NN * 4);
  float* agg  = (float*)(ws + off); off = align_up(off + (size_t)NN * D * 4);
  float* h    = (float*)(ws + off); off = align_up(off + (size_t)NN * D * 4);
  bf16*  Ab   = (bf16*)(ws + off);  off = align_up(off + (size_t)NN * D * 2);
  bf16*  Rb   = (bf16*)(ws + off);  off = align_up(off + (size_t)NN * D * 2);
  bf16*  W1lb = (bf16*)(ws + off);  off = align_up(off + (size_t)D * D * 2);
  bf16*  W1rb = (bf16*)(ws + off);  off = align_up(off + (size_t)D * D * 2);
  bf16*  W2lb = (bf16*)(ws + off);  off = align_up(off + (size_t)D * D * 2);
  bf16*  W2rb = (bf16*)(ws + off);  off = align_up(off + (size_t)D * D * 2);
  float* gsum = (float*)(ws + off); off = align_up(off + (size_t)(NG * D + NG) * 4);
  float* gcnt = gsum + (size_t)NG * D;

  const int T = 256;
  const int nd = NN * D;                       // 12.8M elements
  dim3 blk(T);
  dim3 gElem((nd + T - 1) / T);
  dim3 gNode((NN + T - 1) / T);
  dim3 gEdge((NE + T - 1) / T);
  dim3 gScat((int)(((long long)NE * 32 + T - 1) / T));
  dim3 gPool((int)(((long long)NN * 32 + T - 1) / T));
  dim3 gW((D * D + T - 1) / T);
  dim3 gGemm((NN / 16 + (T / 32) - 1) / (T / 32));   // 6250 waves -> 782 blocks
  dim3 gPoolZ((NG * D + NG + T - 1) / T);
  dim3 gFin((NG * NC + T - 1) / T);

  // ---- prep: zeros + bf16 weight/feature conversion
  zero_f32<<<gNode, blk, 0, stream>>>(deg, NN);
  zero_f32<<<gElem, blk, 0, stream>>>(agg, nd);
  zero_f32<<<gPoolZ, blk, 0, stream>>>(gsum, NG * D + NG);
  f2bf_kernel<<<gW, blk, 0, stream>>>(W1l, W1lb, D * D);
  f2bf_kernel<<<gW, blk, 0, stream>>>(W1r, W1rb, D * D);
  f2bf_kernel<<<gW, blk, 0, stream>>>(W2l, W2lb, D * D);
  f2bf_kernel<<<gW, blk, 0, stream>>>(W2r, W2rb, D * D);
  f2bf_kernel<<<gElem, blk, 0, stream>>>(x, Rb, nd);   // Rb = bf16(x)

  // ---- degree
  deg_count<<<gEdge, blk, 0, stream>>>(dst, deg);
  deg_invert<<<gNode, blk, 0, stream>>>(deg);          // deg := 1/max(deg,1)

  // ---- layer 1
  scatter_add<<<gScat, blk, 0, stream>>>(src, dst, x, agg);
  scale_to_bf16<<<gElem, blk, 0, stream>>>(agg, deg, Ab);
  sage_gemm<<<gGemm, blk, 0, stream>>>(Ab, Rb, W1lb, W1rb, b1, h, NN);

  // ---- layer 2 (reuse buffers)
  zero_f32<<<gElem, blk, 0, stream>>>(agg, nd);
  f2bf_kernel<<<gElem, blk, 0, stream>>>(h, Rb, nd);   // Rb = bf16(h)
  scatter_add<<<gScat, blk, 0, stream>>>(src, dst, h, agg);
  scale_to_bf16<<<gElem, blk, 0, stream>>>(agg, deg, Ab);
  sage_gemm<<<gGemm, blk, 0, stream>>>(Ab, Rb, W2lb, W2rb, b2, h, NN);

  // ---- global mean pool + head
  pool_sum<<<gPool, blk, 0, stream>>>(h, batch, gsum);
  pool_cnt<<<gNode, blk, 0, stream>>>(batch, gcnt);
  final_lin<<<gFin, blk, 0, stream>>>(gsum, gcnt, Wlin, blin, out);
}
